// GAT_8847632630371
// MI455X (gfx1250) — compile-verified, matrix operand order checked by reading, not run on hardware
//
#include <hip/hip_runtime.h>
#include <hip/hip_bf16.h>
#include <stdint.h>

#define D 128
#define NEG_SLOPE 0.01f

typedef __attribute__((ext_vector_type(16))) __bf16 v16bf;
typedef __attribute__((ext_vector_type(8)))  float  v8f;

union Frag {
    v16bf bf;
    unsigned int ui[8];
    uint4 u4[2];
};

__device__ __forceinline__ unsigned int pack_bf16(float a, float b) {
    unsigned int ua = __float_as_uint(a), ub = __float_as_uint(b);
    ua = (ua + 0x7FFFu + ((ua >> 16) & 1u)) >> 16;   // round-to-nearest-even
    ub = (ub + 0x7FFFu + ((ub >> 16) & 1u)) >> 16;
    return (ua & 0xFFFFu) | (ub << 16);
}

// ---------------- init: segment-max = -inf, denom = 0 ----------------
__global__ void gat_init(float* __restrict__ mbuf, float* __restrict__ denom, int n) {
    int i = blockIdx.x * blockDim.x + threadIdx.x;
    if (i < n) { mbuf[i] = -__builtin_huge_valf(); denom[i] = 0.0f; }
}

// ---------------- W -> W^T in bf16 (Wt[n*D + k] = bf16(W[k*D + n])) ----------------
__global__ void gat_wt(const float* __restrict__ W, unsigned short* __restrict__ Wt) {
    int i = blockIdx.x * blockDim.x + threadIdx.x;
    if (i < D * D) {
        int nIdx = i / D, k = i % D;
        unsigned int u = __float_as_uint(W[(size_t)k * D + nIdx]);
        Wt[i] = (unsigned short)((u + 0x7FFFu + ((u >> 16) & 1u)) >> 16);
    }
}

// ---------------- z = h @ W via v_wmma_f32_16x16x32_bf16 ----------------
// one wave per 16-row tile; 8 accumulator tiles (16x128 output) per wave
__global__ __launch_bounds__(256) void gat_gemm_wmma(
    const float* __restrict__ h, const unsigned short* __restrict__ Wt,
    float* __restrict__ z, int n_tiles) {
    const int wave = threadIdx.x >> 5;
    const int tile = blockIdx.x * 8 + wave;
    if (tile >= n_tiles) return;
    const int lane = threadIdx.x & 31;
    const int nl   = lane & 15;   // A: row M within tile;  B/D: col N within tile
    const int half = lane >> 4;

    const float* hrow = h + (size_t)(tile * 16 + nl) * D;

    v8f acc[8];
#pragma unroll
    for (int i = 0; i < 8; ++i) acc[i] = v8f{0.f, 0.f, 0.f, 0.f, 0.f, 0.f, 0.f, 0.f};

#pragma unroll
    for (int kb = 0; kb < D; kb += 32) {
        // A fragment: 16 bf16 per lane = two contiguous 8-float chunks, cvt on the fly
        Frag a;
        const float4 c0 = *(const float4*)(hrow + kb + half * 8);
        const float4 c1 = *(const float4*)(hrow + kb + half * 8 + 4);
        const float4 c2 = *(const float4*)(hrow + kb + 16 + half * 8);
        const float4 c3 = *(const float4*)(hrow + kb + 16 + half * 8 + 4);
        a.ui[0] = pack_bf16(c0.x, c0.y);
        a.ui[1] = pack_bf16(c0.z, c0.w);
        a.ui[2] = pack_bf16(c1.x, c1.y);
        a.ui[3] = pack_bf16(c1.z, c1.w);
        a.ui[4] = pack_bf16(c2.x, c2.y);
        a.ui[5] = pack_bf16(c2.z, c2.w);
        a.ui[6] = pack_bf16(c3.x, c3.y);
        a.ui[7] = pack_bf16(c3.z, c3.w);
#pragma unroll
        for (int nt = 0; nt < 8; ++nt) {
            // B fragment: lane holds column (nt*16 + nl) of W = row of Wt, same K order
            Frag b;
            const unsigned short* wrow = Wt + (size_t)(nt * 16 + nl) * D;
            b.u4[0] = *(const uint4*)(wrow + kb + half * 8);
            b.u4[1] = *(const uint4*)(wrow + kb + 16 + half * 8);
            acc[nt] = __builtin_amdgcn_wmma_f32_16x16x32_bf16(
                false, a.bf, false, b.bf, (short)0, acc[nt], false, false);
        }
    }

    // C/D layout: VGPR r -> row half*8 + r, col = nt*16 + nl
    float* zt = z + (size_t)tile * 16 * D;
#pragma unroll
    for (int nt = 0; nt < 8; ++nt)
#pragma unroll
        for (int r = 0; r < 8; ++r)
            zt[(half * 8 + r) * D + nt * 16 + nl] = acc[nt][r];
}

// ---------------- scalar tail for rows not covered by 16-row tiles ----------------
__global__ void gat_gemm_tail(const float* __restrict__ h, const float* __restrict__ W,
                              float* __restrict__ z, int row0, int n) {
    int idx = blockIdx.x * blockDim.x + threadIdx.x;
    int row = row0 + idx / D;
    int col = idx % D;
    if (row >= n) return;
    float s = 0.f;
    for (int k = 0; k < D; ++k) s += h[(size_t)row * D + k] * W[(size_t)k * D + col];
    z[(size_t)row * D + col] = s;
}

// ---------------- s_src / s_dst: wave-per-node dot products ----------------
__global__ __launch_bounds__(256) void gat_scores(
    const float* __restrict__ z, const float* __restrict__ attn,
    float* __restrict__ s_src, float* __restrict__ s_dst, int n) {
    const int wave = threadIdx.x >> 5;
    const int node = blockIdx.x * 8 + wave;
    if (node >= n) return;
    const int lane = threadIdx.x & 31;
    const float4 zv = *(const float4*)(z + (size_t)node * D + lane * 4);
    const float4 as = *(const float4*)(attn + lane * 4);
    const float4 ad = *(const float4*)(attn + D + lane * 4);
    float ss = zv.x * as.x + zv.y * as.y + zv.z * as.z + zv.w * as.w;
    float sd = zv.x * ad.x + zv.y * ad.y + zv.z * ad.z + zv.w * ad.w;
#pragma unroll
    for (int o = 16; o > 0; o >>= 1) {
        ss += __shfl_xor(ss, o, 32);
        sd += __shfl_xor(sd, o, 32);
    }
    if (lane == 0) { s_src[node] = ss; s_dst[node] = sd; }
}

// ---------------- float atomic max via sign-magnitude int punning ----------------
__device__ __forceinline__ void atomicMaxF(float* addr, float v) {
    if (v >= 0.0f) atomicMax((int*)addr, __float_as_int(v));
    else           atomicMin((unsigned int*)addr, __float_as_uint(v));
}

// ---------------- e = leaky_relu(s_src[src] + s_dst[dst]); segment max ----------------
__global__ void gat_edge_logit(const int* __restrict__ src, const int* __restrict__ dst,
                               const float* __restrict__ s_src, const float* __restrict__ s_dst,
                               float* __restrict__ ebuf, float* __restrict__ mbuf, int ne) {
    int i = blockIdx.x * blockDim.x + threadIdx.x;
    if (i >= ne) return;
    float v = s_src[src[i]] + s_dst[dst[i]];
    float e = v > 0.0f ? v : NEG_SLOPE * v;
    ebuf[i] = e;
    atomicMaxF(&mbuf[dst[i]], e);
}

// ---------------- ex = exp(e - m[dst]); segment sum of denominators ----------------
__global__ void gat_edge_exp(const int* __restrict__ dst, float* __restrict__ ebuf,
                             const float* __restrict__ mbuf, float* __restrict__ denom, int ne) {
    int i = blockIdx.x * blockDim.x + threadIdx.x;
    if (i >= ne) return;
    int d = dst[i];
    float ex = __expf(ebuf[i] - mbuf[d]);
    ebuf[i] = ex;
    atomicAdd(&denom[d], ex);
}

// ---------------- out[dst] += (ex/denom[dst]) * z[src]; wave-per-edge ----------------
__global__ __launch_bounds__(256) void gat_scatter(
    const int* __restrict__ src, const int* __restrict__ dst,
    const float* __restrict__ ebuf, const float* __restrict__ denom,
    const float* __restrict__ z, float* __restrict__ out, int ne) {
    const int wave = threadIdx.x >> 5;
    const int i = blockIdx.x * 8 + wave;
    if (i >= ne) return;
    const int lane = threadIdx.x & 31;
    const int s = src[i], d = dst[i];
    const float alpha = ebuf[i] / denom[d];
    const float4 zv = *(const float4*)(z + (size_t)s * D + lane * 4);
    float* o = out + (size_t)d * D + lane * 4;
    atomicAdd(o + 0, alpha * zv.x);
    atomicAdd(o + 1, alpha * zv.y);
    atomicAdd(o + 2, alpha * zv.z);
    atomicAdd(o + 3, alpha * zv.w);
}

extern "C" void kernel_launch(void* const* d_in, const int* in_sizes, int n_in,
                              void* d_out, int out_size, void* d_ws, size_t ws_size,
                              hipStream_t stream) {
    const float* h    = (const float*)d_in[0];
    const float* W    = (const float*)d_in[1];
    const float* attn = (const float*)d_in[2];
    const int*   esrc = (const int*)d_in[3];
    const int*   edst = (const int*)d_in[4];
    const int n  = in_sizes[0] / D;
    const int ne = in_sizes[3];

    char* ws = (char*)d_ws;
    size_t off = 0;
    auto alloc = [&](size_t bytes) -> void* {
        void* p = ws + off;
        off = (off + bytes + 255) & ~(size_t)255;
        return p;
    };
    float*          z     = (float*)alloc((size_t)n * D * sizeof(float));
    unsigned short* Wt    = (unsigned short*)alloc((size_t)D * D * sizeof(unsigned short));
    float*          s_src = (float*)alloc((size_t)n * sizeof(float));
    float*          s_dst = (float*)alloc((size_t)n * sizeof(float));
    float*          mbuf  = (float*)alloc((size_t)n * sizeof(float));
    float*          denom = (float*)alloc((size_t)n * sizeof(float));
    float*          ebuf  = (float*)alloc((size_t)ne * sizeof(float));
    (void)ws_size;

    hipMemsetAsync(d_out, 0, (size_t)out_size * sizeof(float), stream);
    gat_init<<<(n + 255) / 256, 256, 0, stream>>>(mbuf, denom, n);
    gat_wt<<<(D * D + 255) / 256, 256, 0, stream>>>(W, Wt);

    const int n_tiles = n / 16;
    if (n_tiles > 0)
        gat_gemm_wmma<<<(n_tiles + 7) / 8, 256, 0, stream>>>(h, Wt, z, n_tiles);
    const int row0 = n_tiles * 16;
    const int rem  = n - row0;
    if (rem > 0)
        gat_gemm_tail<<<(rem * D + 255) / 256, 256, 0, stream>>>(h, W, z, row0, n);

    gat_scores<<<(n + 7) / 8, 256, 0, stream>>>(z, attn, s_src, s_dst, n);
    gat_edge_logit<<<(ne + 255) / 256, 256, 0, stream>>>(esrc, edst, s_src, s_dst, ebuf, mbuf, ne);
    gat_edge_exp<<<(ne + 255) / 256, 256, 0, stream>>>(edst, ebuf, mbuf, denom, ne);
    gat_scatter<<<(ne + 7) / 8, 256, 0, stream>>>(esrc, edst, ebuf, denom, z, (float*)d_out, ne);
}
